// AttentionEncoder_56006373540233
// MI455X (gfx1250) — matrix operand order repaired and checked by reading, not compiled, not run on hardware
//
#include <hip/hip_runtime.h>
#include <hip/hip_bf16.h>
#include <math.h>

// ---------------------------------------------------------------------------
// Problem constants (B=8, L=1024, D=1024, H=8, DH=128, VOCAB=32000)
// SCALE = (DH // H) ** -0.5 = 16 ** -0.5 = 0.25  (faithful to reference bug)
// q.reshape(B*H, L, DH) is a flat reinterpretation: slab n (0..63) of the
// [8192,1024] activation buffer is the contiguous 1024x128 block at offset
// n*131072.  Mask row for slab n is tokens[(n % 8), :].
// ---------------------------------------------------------------------------

typedef __attribute__((ext_vector_type(16))) __bf16 v16bf;
typedef __attribute__((ext_vector_type(8)))  float  v8f;

#define NEG_INF (-__builtin_inff())

// ---- WMMA fragment loaders (bf16, 16x16x32) --------------------------------
// A (16x32, MxK) row-major source, leading dim ld:
//   lane: row = lane&15, half = lane>>4;  k(i) = 8*half + (i%8) + (i/8)*16
__device__ __forceinline__ v16bf load_a_frag(const __bf16* base, int ld) {
  const int lane = threadIdx.x & 31;
  const int row  = lane & 15;
  const int half = lane >> 4;
  const __bf16* p = base + row * ld + half * 8;
  v16bf a;
#pragma unroll
  for (int i = 0; i < 8; ++i) { a[i] = p[i]; a[8 + i] = p[16 + i]; }
  return a;
}

// B (32x16, KxN) supplied as Bt[n][k] row-major (i.e. transposed), ld over k:
//   lane: col = lane&15, half = lane>>4;  k(i) = 8*half + (i%8) + (i/8)*16
__device__ __forceinline__ v16bf load_b_frag_t(const __bf16* base, int ld) {
  const int lane = threadIdx.x & 31;
  const int col  = lane & 15;
  const int half = lane >> 4;
  const __bf16* p = base + col * ld + half * 8;
  v16bf b;
#pragma unroll
  for (int i = 0; i < 8; ++i) { b[i] = p[i]; b[8 + i] = p[16 + i]; }
  return b;
}

__device__ __forceinline__ v8f wmma_bf16(v16bf a, v16bf b, v8f c) {
  return __builtin_amdgcn_wmma_f32_16x16x32_bf16(false, a, false, b, (short)0, c,
                                                 false, false);
}

// ---------------------------------------------------------------------------
// 1) Embedding gather:  x[t,d] = emb[tokens[t], d]   (fp32 + bf16 copies)
// ---------------------------------------------------------------------------
__global__ void gather_kernel(const int* __restrict__ tokens,
                              const float* __restrict__ emb,
                              float* __restrict__ xf, __bf16* __restrict__ xb) {
  const int t   = blockIdx.x;                 // 0..8191
  const int tok = tokens[t];
#pragma unroll
  for (int j = 0; j < 4; ++j) {
    const int d = threadIdx.x + j * 256;
    const float v = emb[tok * 1024 + d];
    xf[t * 1024 + d] = v;
    xb[t * 1024 + d] = (__bf16)v;
  }
}

// ---------------------------------------------------------------------------
// 2) Weight transpose + bf16 convert: Wt[n,k] = W[k,n]   (1024x1024)
// ---------------------------------------------------------------------------
__global__ void transpose_w_kernel(const float* __restrict__ W,
                                   __bf16* __restrict__ Wt) {
  const int idx = blockIdx.x * 256 + threadIdx.x;  // over 1M
  const int n = idx >> 10, k = idx & 1023;
  Wt[idx] = (__bf16)W[k * 1024 + n];
}

// ---------------------------------------------------------------------------
// 3) WMMA GEMM:  Y = A @ W + bias   (A: MxK bf16 row-major, Wt: NxK bf16)
//    wave tile: 64x64 (4x4 16x16 tiles) -> 16 WMMA per 32-K step against
//    16 b128 loads.  block = 8 waves -> 64 x 512.  grid = (M/64, N/512).
//    Output mode is a compile-time template -> branch-free epilogue.
// ---------------------------------------------------------------------------
template <bool OUT_F32, bool OUT_BF16>
__global__ __launch_bounds__(256, 1)
void gemm_bf16_kernel(const __bf16* __restrict__ A,
                      const __bf16* __restrict__ Bt,
                      const float* __restrict__ bias,
                      float* __restrict__ Cf, __bf16* __restrict__ Cb,
                      int M, int N, int K) {
  const int wave = threadIdx.x >> 5;
  const int lane = threadIdx.x & 31;
  const int half = lane >> 4;
  const int col  = lane & 15;
  const int m0 = blockIdx.x * 64;
  const int n0 = (blockIdx.y * 8 + wave) * 64;

  v8f acc[4][4];
  v8f z = {};
#pragma unroll
  for (int i = 0; i < 4; ++i)
#pragma unroll
    for (int j = 0; j < 4; ++j) acc[i][j] = z;

  for (int k = 0; k < K; k += 32) {
    v16bf a[4];
#pragma unroll
    for (int i = 0; i < 4; ++i)
      a[i] = load_a_frag(A + (m0 + i * 16) * K + k, K);
#pragma unroll
    for (int j = 0; j < 4; ++j) {
      v16bf b = load_b_frag_t(Bt + (n0 + j * 16) * K + k, K);
#pragma unroll
      for (int i = 0; i < 4; ++i) acc[i][j] = wmma_bf16(a[i], b, acc[i][j]);
    }
  }

#pragma unroll
  for (int i = 0; i < 4; ++i)
#pragma unroll
    for (int j = 0; j < 4; ++j) {
      const int c  = n0 + j * 16 + col;
      const float bc = bias[c];
      const int row0 = m0 + i * 16 + 8 * half;
#pragma unroll
      for (int r = 0; r < 8; ++r) {
        const float v = acc[i][j][r] + bc;
        if constexpr (OUT_F32)  Cf[(row0 + r) * N + c] = v;
        if constexpr (OUT_BF16) Cb[(row0 + r) * N + c] = (__bf16)v;
      }
    }
}

// ---------------------------------------------------------------------------
// 4) V transpose per slab: Vt[n][dh,key] = V_flat[n][key,dh]
//    flat idx f -> slab n=f>>17, l=(f&131071)>>7, dh=f&127
// ---------------------------------------------------------------------------
__global__ void transpose_v_kernel(const __bf16* __restrict__ v,
                                   __bf16* __restrict__ vt) {
  const int f = blockIdx.x * 256 + threadIdx.x;  // over 8M
  const int n = f >> 17, rem = f & 131071;
  const int l = rem >> 7, dh = rem & 127;
  vt[n * 131072 + dh * 1024 + l] = v[f];
}

// ---------------------------------------------------------------------------
// 5) Flash attention over slabs: out = softmax(mask(Q K^T * 0.25)) V
//    grid (64 slabs, 8), block 256 (8 waves).  Each wave: 16 query rows,
//    streams keys in tiles of 32.  P-tile staged through LDS to convert
//    C-layout f32 -> A-layout bf16 for the P @ V WMMA.
//    launch_bounds(256,1): keep qa + acc resident, no scratch spills.
// ---------------------------------------------------------------------------
__global__ __launch_bounds__(256, 1)
void attention_kernel(const __bf16* __restrict__ qg,
                      const __bf16* __restrict__ kg,
                      const __bf16* __restrict__ vtg,
                      const int* __restrict__ tokens,
                      __bf16* __restrict__ ctx) {
  const int n    = blockIdx.x;            // slab 0..63
  const int wave = threadIdx.x >> 5;
  const int lane = threadIdx.x & 31;
  const int half = lane >> 4;
  const int nn   = lane & 15;
  const int q0   = (blockIdx.y * 8 + wave) * 16;

  const __bf16* Q  = qg  + n * 131072;    // [1024,128] row-major
  const __bf16* Ks = kg  + n * 131072;    // [1024,128] row-major
  const __bf16* Vt = vtg + n * 131072;    // [128,1024] dh-major
  const int* tok = tokens + (n & 7) * 1024;

  __shared__ __bf16 pstage[8][16 * 32];

  v16bf qa[4];
#pragma unroll
  for (int d = 0; d < 4; ++d) qa[d] = load_a_frag(Q + q0 * 128 + d * 32, 128);

  v8f acc[8];
  v8f z = {};
#pragma unroll
  for (int f = 0; f < 8; ++f) acc[f] = z;
  float mrow[8], lrow[8];
#pragma unroll
  for (int r = 0; r < 8; ++r) { mrow[r] = NEG_INF; lrow[r] = 0.0f; }

  for (int kb = 0; kb < 1024; kb += 32) {
    // ---- scores: 16 queries x 32 keys, fp32 accum -------------------------
    v8f s0 = z, s1 = z;
#pragma unroll
    for (int d = 0; d < 4; ++d) {
      v16bf b0 = load_b_frag_t(Ks + (kb)      * 128 + d * 32, 128);
      v16bf b1 = load_b_frag_t(Ks + (kb + 16) * 128 + d * 32, 128);
      s0 = wmma_bf16(qa[d], b0, s0);
      s1 = wmma_bf16(qa[d], b1, s1);
    }
    const bool pad0 = (tok[kb + nn] == 0);
    const bool pad1 = (tok[kb + 16 + nn] == 0);
#pragma unroll
    for (int r = 0; r < 8; ++r) {
      s0[r] = pad0 ? NEG_INF : s0[r] * 0.25f;
      s1[r] = pad1 ? NEG_INF : s1[r] * 0.25f;
    }
    // ---- online softmax (row reduce across 16 lanes of this half) ---------
    float mt[8];
#pragma unroll
    for (int r = 0; r < 8; ++r) mt[r] = fmaxf(s0[r], s1[r]);
#pragma unroll
    for (int off = 1; off < 16; off <<= 1)
#pragma unroll
      for (int r = 0; r < 8; ++r) mt[r] = fmaxf(mt[r], __shfl_xor(mt[r], off, 32));

    float corr[8];
#pragma unroll
    for (int r = 0; r < 8; ++r) {
      const float mnew = fmaxf(mrow[r], mt[r]);
      corr[r] = (mnew == NEG_INF) ? 1.0f : __expf(mrow[r] - mnew);
      mrow[r] = mnew;
    }
    float ps[8];
#pragma unroll
    for (int r = 0; r < 8; ++r) {
      const float p0 = (mrow[r] == NEG_INF) ? 0.0f : __expf(s0[r] - mrow[r]);
      const float p1 = (mrow[r] == NEG_INF) ? 0.0f : __expf(s1[r] - mrow[r]);
      s0[r] = p0; s1[r] = p1;
      ps[r] = p0 + p1;
    }
#pragma unroll
    for (int off = 1; off < 16; off <<= 1)
#pragma unroll
      for (int r = 0; r < 8; ++r) ps[r] += __shfl_xor(ps[r], off, 32);
#pragma unroll
    for (int r = 0; r < 8; ++r) lrow[r] = lrow[r] * corr[r] + ps[r];
#pragma unroll
    for (int f = 0; f < 8; ++f)
#pragma unroll
      for (int r = 0; r < 8; ++r) acc[f][r] *= corr[r];

    // ---- P tile: C layout (col nn, rows r+8*half) -> LDS row-major 16x32 --
#pragma unroll
    for (int r = 0; r < 8; ++r) {
      pstage[wave][(r + 8 * half) * 32 + nn]      = (__bf16)s0[r];
      pstage[wave][(r + 8 * half) * 32 + 16 + nn] = (__bf16)s1[r];
    }
    // reload as A-frag (wave-local; DS ops are in-order within a wave)
    v16bf pa;
#pragma unroll
    for (int i = 0; i < 8; ++i) {
      pa[i]     = pstage[wave][nn * 32 + half * 8 + i];
      pa[8 + i] = pstage[wave][nn * 32 + half * 8 + 16 + i];
    }
    // ---- P @ V : 8 dh tiles of 16 ----------------------------------------
#pragma unroll
    for (int f = 0; f < 8; ++f) {
      v16bf bv = load_b_frag_t(Vt + (f * 16) * 1024 + kb, 1024);
      acc[f] = wmma_bf16(pa, bv, acc[f]);
    }
  }

  // ---- epilogue: ctx[n][q,dh] = acc / l ----------------------------------
#pragma unroll
  for (int f = 0; f < 8; ++f)
#pragma unroll
    for (int r = 0; r < 8; ++r) {
      const int qrow = q0 + r + 8 * half;
      const int dh   = f * 16 + nn;
      ctx[n * 131072 + qrow * 128 + dh] = (__bf16)(acc[f][r] / lrow[r]);
    }
}

// ---------------------------------------------------------------------------
// 6) Residual + max/mean pool:  agg[b, 0:1024]=max_l(x+y), agg[b,1024:]=mean
// ---------------------------------------------------------------------------
__global__ void residual_pool_kernel(const float* __restrict__ x,
                                     const float* __restrict__ y,
                                     float* __restrict__ agg) {
  const int d = blockIdx.x * 256 + threadIdx.x;  // 0..1023
  const int b = blockIdx.y;
  float mx = NEG_INF, sm = 0.0f;
  for (int l = 0; l < 1024; ++l) {
    const int idx = (b * 1024 + l) * 1024 + d;
    const float r = x[idx] + y[idx];
    mx = fmaxf(mx, r);
    sm += r;
  }
  agg[b * 2048 + d]        = mx;
  agg[b * 2048 + 1024 + d] = sm * (1.0f / 1024.0f);
}

// ---------------------------------------------------------------------------
// 7) LayerNorm over 2048 per batch row
// ---------------------------------------------------------------------------
__global__ void layernorm_kernel(const float* __restrict__ agg,
                                 const float* __restrict__ gamma,
                                 const float* __restrict__ beta,
                                 float* __restrict__ out) {
  const int b = blockIdx.x;
  __shared__ float red[2][8];
  float s = 0.0f, s2 = 0.0f;
  for (int j = threadIdx.x; j < 2048; j += 256) {
    const float v = agg[b * 2048 + j];
    s += v; s2 += v * v;
  }
#pragma unroll
  for (int off = 16; off > 0; off >>= 1) {
    s  += __shfl_xor(s, off, 32);
    s2 += __shfl_xor(s2, off, 32);
  }
  const int wave = threadIdx.x >> 5;
  if ((threadIdx.x & 31) == 0) { red[0][wave] = s; red[1][wave] = s2; }
  __syncthreads();
  float ts = 0.0f, ts2 = 0.0f;
#pragma unroll
  for (int w = 0; w < 8; ++w) { ts += red[0][w]; ts2 += red[1][w]; }
  const float mu  = ts * (1.0f / 2048.0f);
  const float var = ts2 * (1.0f / 2048.0f) - mu * mu;
  const float inv = rsqrtf(var + 1e-5f);
  for (int j = threadIdx.x; j < 2048; j += 256)
    out[b * 2048 + j] = (agg[b * 2048 + j] - mu) * inv * gamma[j] + beta[j];
}

// ---------------------------------------------------------------------------
// Launch
// ---------------------------------------------------------------------------
extern "C" void kernel_launch(void* const* d_in, const int* in_sizes, int n_in,
                              void* d_out, int out_size, void* d_ws, size_t ws_size,
                              hipStream_t stream) {
  const int*   tokens = (const int*)d_in[0];
  const float* emb    = (const float*)d_in[1];
  const float* Wq     = (const float*)d_in[2];
  const float* bq     = (const float*)d_in[3];
  const float* Wk     = (const float*)d_in[4];
  const float* bk     = (const float*)d_in[5];
  const float* Wv     = (const float*)d_in[6];
  const float* bv     = (const float*)d_in[7];
  const float* Wo     = (const float*)d_in[8];
  const float* bo     = (const float*)d_in[9];
  const float* gamma  = (const float*)d_in[10];
  const float* beta   = (const float*)d_in[11];
  float* out = (float*)d_out;

  char* w = (char*)d_ws;
  float*  xf   = (float*)w;  w += 8192u * 1024u * 4u;       // 32 MB fp32 x
  __bf16* xb   = (__bf16*)w; w += 8192u * 1024u * 2u;       // 16 MB bf16 x
  __bf16* wqt  = (__bf16*)w; w += 1024u * 1024u * 2u;
  __bf16* wkt  = (__bf16*)w; w += 1024u * 1024u * 2u;
  __bf16* wvt  = (__bf16*)w; w += 1024u * 1024u * 2u;
  __bf16* wot  = (__bf16*)w; w += 1024u * 1024u * 2u;
  __bf16* qb   = (__bf16*)w; w += 8192u * 1024u * 2u;
  __bf16* kb2  = (__bf16*)w; w += 8192u * 1024u * 2u;
  __bf16* vb   = (__bf16*)w; w += 8192u * 1024u * 2u;
  __bf16* vt   = (__bf16*)w; w += 8192u * 1024u * 2u;
  __bf16* ctxb = (__bf16*)w; w += 8192u * 1024u * 2u;
  float*  yo   = (float*)w;  w += 8192u * 1024u * 4u;       // 32 MB fp32 ctx@Wo
  float*  agg  = (float*)w;  w += 8u * 2048u * 4u;

  gather_kernel<<<8192, 256, 0, stream>>>(tokens, emb, xf, xb);

  transpose_w_kernel<<<4096, 256, 0, stream>>>(Wq, wqt);
  transpose_w_kernel<<<4096, 256, 0, stream>>>(Wk, wkt);
  transpose_w_kernel<<<4096, 256, 0, stream>>>(Wv, wvt);
  transpose_w_kernel<<<4096, 256, 0, stream>>>(Wo, wot);

  const dim3 gg(128, 2), gb(256);
  gemm_bf16_kernel<false, true><<<gg, gb, 0, stream>>>(
      xb, wqt, bq, (float*)nullptr, qb, 8192, 1024, 1024);
  gemm_bf16_kernel<false, true><<<gg, gb, 0, stream>>>(
      xb, wkt, bk, (float*)nullptr, kb2, 8192, 1024, 1024);
  gemm_bf16_kernel<false, true><<<gg, gb, 0, stream>>>(
      xb, wvt, bv, (float*)nullptr, vb, 8192, 1024, 1024);

  transpose_v_kernel<<<32768, 256, 0, stream>>>(vb, vt);

  attention_kernel<<<dim3(64, 8), 256, 0, stream>>>(qb, kb2, vt, tokens, ctxb);

  gemm_bf16_kernel<true, false><<<gg, gb, 0, stream>>>(
      ctxb, wot, bo, yo, (__bf16*)nullptr, 8192, 1024, 1024);

  residual_pool_kernel<<<dim3(4, 8), 256, 0, stream>>>(xf, yo, agg);
  layernorm_kernel<<<8, 256, 0, stream>>>(agg, gamma, beta, out);
}
